// Encoder_2860448219141
// MI455X (gfx1250) — matrix operand order repaired and checked by reading, not compiled, not run on hardware
//
// Performer (FAVOR+) encoder layer for MI455X (gfx1250), bf16 WMMA + f32 accum.
//
// Roofline: ~450 GFLOP of GEMM work vs <1 GB HBM traffic @ 23.3 TB/s -> matrix-
// core bound. All GEMMs run on v_wmma_f32_16x16x32_bf16 (wave32, 16x16 tiles).
// GEMM uses a double-buffered global->VGPR->LDS pipeline (one barrier per
// K-step, global latency hidden under the WMMA stream). Weights pre-transposed
// so every large GEMM operand stages with b128 loads/stores. FAVOR feature maps
// fuse exp/max/diag epilogues in-register so fp32 data_dash never touches HBM.

#include <hip/hip_runtime.h>
#include <math.h>

typedef __attribute__((ext_vector_type(16))) __bf16 v16bf;
typedef __attribute__((ext_vector_type(8)))  __bf16 v8bf;
typedef __attribute__((ext_vector_type(8)))  float  v8f;

union BFrag { v16bf v; v8bf h[2]; };

__device__ __forceinline__ v8f vzero8() {
  v8f z;
#pragma unroll
  for (int i = 0; i < 8; ++i) z[i] = 0.0f;
  return z;
}
__device__ __forceinline__ v8bf bzero8() {
  v8bf z;
#pragma unroll
  for (int i = 0; i < 8; ++i) z[i] = (__bf16)0.0f;
  return z;
}

__device__ __forceinline__ v8f wmma_bf16(v16bf a, v16bf b, v8f c) {
  return __builtin_amdgcn_wmma_f32_16x16x32_bf16(false, a, false, b, (short)0, c,
                                                 false, false);
}

// monotonic uint key for float atomic-max
__device__ __forceinline__ unsigned fkey(float f) {
  unsigned u = __float_as_uint(f);
  return (u & 0x80000000u) ? ~u : (u | 0x80000000u);
}
__device__ __forceinline__ float funkey(unsigned k) {
  return __uint_as_float((k & 0x80000000u) ? (k & 0x7FFFFFFFu) : ~k);
}

// ---------------------------------------------------------------------------
// Generic batched strided GEMM: C[z] = act(A[z] * B[z] + bias)
// A(i,k) = A[i*sAr + k*sAc]; B(k,j) = B[k*sBr + j*sBc].  Kdim % 32 == 0.
// Staging: K-contiguous (sAc==1 / sBr==1) -> b128 load + b128 LDS store;
// else M/N-contiguous -> b128 load + 8x b16 LDS stores (transpose).
// Double-buffered LDS: next tile loads to VGPRs during WMMA on current tile.
// C(i,j) stored at i*ldcr + j*ldcc (allows transposed output).
// Block tile 128x128x32, 8 waves, each wave 32x64 via 2x4 wmma 16x16 tiles.
// Batch offset: z -> (z/hmod)*outer + (z%hmod)*inner.
// ---------------------------------------------------------------------------
__global__ __launch_bounds__(256) void gemm_wmma(
    const __bf16* __restrict__ A, long long sAr, long long sAc, long long sAb0,
    long long sAb1, const __bf16* __restrict__ Bm, long long sBr, long long sBc,
    long long sBb0, long long sBb1, const float* __restrict__ bias,
    float* __restrict__ Cf, __bf16* __restrict__ Cb, long long ldcr,
    long long ldcc, long long sCb0, long long sCb1, int Mdim, int Ndim,
    int Kdim, int hmod, int act) {
  constexpr int BM = 128, BN = 128, BK = 32, LDSS = 40;
  __shared__ __align__(16) __bf16 As[2][BM * LDSS];
  __shared__ __align__(16) __bf16 Bs[2][BN * LDSS];

  const int z = blockIdx.z;
  const int zo = z / hmod, zi = z % hmod;
  const __bf16* Ab = A + (long long)zo * sAb0 + (long long)zi * sAb1;
  const __bf16* Bb = Bm + (long long)zo * sBb0 + (long long)zi * sBb1;
  const long long coff = (long long)zo * sCb0 + (long long)zi * sCb1;

  const int m0 = blockIdx.y * BM, n0 = blockIdx.x * BN;
  const int tid = threadIdx.x, lane = tid & 31, wave = tid >> 5;
  const int wm0 = (wave >> 1) * 32, wn0 = (wave & 1) * 64;
  const int half = lane >> 4, l15 = lane & 15;
  const bool aK = (sAc == 1), bK = (sBr == 1);

  // per-thread staging coordinates (2 groups of 8 elements per operand)
  int rA[2], cA[2], cAT[2], m8A[2];
  int rB[2], cB[2], cBT[2], n8B[2];
#pragma unroll
  for (int t = 0; t < 2; ++t) {
    int i = tid + t * 256;
    rA[t] = i >> 2;  cA[t] = (i & 3) * 8;   // K-contiguous layout
    cAT[t] = i >> 4; m8A[t] = (i & 15) * 8; // transposed layout
    rB[t] = rA[t];   cB[t] = cA[t];
    cBT[t] = cAT[t]; n8B[t] = m8A[t];
  }

  auto load_tile = [&](int k0, v8bf aR[2], v8bf bR[2]) {
    if (aK) {
#pragma unroll
      for (int t = 0; t < 2; ++t) {
        int gr = m0 + rA[t];
        aR[t] = (gr < Mdim)
                    ? *(const v8bf*)(Ab + (long long)gr * sAr + k0 + cA[t])
                    : bzero8();
      }
    } else {  // sAr==1: contiguous along M
#pragma unroll
      for (int t = 0; t < 2; ++t) {
        aR[t] = (m0 + m8A[t] + 7 < Mdim)
                    ? *(const v8bf*)(Ab + (long long)(k0 + cAT[t]) * sAc +
                                     (m0 + m8A[t]))
                    : bzero8();
      }
    }
    if (bK) {
#pragma unroll
      for (int t = 0; t < 2; ++t) {
        int gn = n0 + rB[t];
        bR[t] = (gn < Ndim)
                    ? *(const v8bf*)(Bb + (long long)gn * sBc + k0 + cB[t])
                    : bzero8();
      }
    } else {  // sBc==1: contiguous along N
#pragma unroll
      for (int t = 0; t < 2; ++t) {
        bR[t] = (n0 + n8B[t] + 7 < Ndim)
                    ? *(const v8bf*)(Bb + (long long)(k0 + cBT[t]) * sBr +
                                     (n0 + n8B[t]))
                    : bzero8();
      }
    }
  };
  auto store_tile = [&](int buf, const v8bf aR[2], const v8bf bR[2]) {
    if (aK) {
#pragma unroll
      for (int t = 0; t < 2; ++t)
        *(v8bf*)&As[buf][rA[t] * LDSS + cA[t]] = aR[t];
    } else {
#pragma unroll
      for (int t = 0; t < 2; ++t)
#pragma unroll
        for (int e = 0; e < 8; ++e)
          As[buf][(m8A[t] + e) * LDSS + cAT[t]] = aR[t][e];
    }
    if (bK) {
#pragma unroll
      for (int t = 0; t < 2; ++t)
        *(v8bf*)&Bs[buf][rB[t] * LDSS + cB[t]] = bR[t];
    } else {
#pragma unroll
      for (int t = 0; t < 2; ++t)
#pragma unroll
        for (int e = 0; e < 8; ++e)
          Bs[buf][(n8B[t] + e) * LDSS + cBT[t]] = bR[t][e];
    }
  };

  v8f acc[2][4];
#pragma unroll
  for (int mt = 0; mt < 2; ++mt)
#pragma unroll
    for (int nt = 0; nt < 4; ++nt) acc[mt][nt] = vzero8();

  const int nk = Kdim / BK;
  {
    v8bf aR[2], bR[2];
    load_tile(0, aR, bR);
    store_tile(0, aR, bR);
  }
  __syncthreads();

  int buf = 0;
  for (int kt = 0; kt < nk; ++kt) {
    const bool more = (kt + 1 < nk);
    v8bf aN[2], bN[2];
    if (more) load_tile((kt + 1) * BK, aN, bN);
    if (kt + 2 < nk) {  // warm L2 two tiles ahead (global_prefetch_b8)
      long long ka = (long long)(kt + 2) * BK;
      __builtin_prefetch(Ab + (long long)(m0 + (tid & 127)) * sAr + ka * sAc, 0, 1);
      __builtin_prefetch(Bb + ka * sBr + (long long)(n0 + (tid & 127)) * sBc, 0, 1);
    }

    const __bf16* asb = As[buf];
    const __bf16* bsb = Bs[buf];
    BFrag af[2];
#pragma unroll
    for (int mt = 0; mt < 2; ++mt) {
      const __bf16* p = &asb[(wm0 + mt * 16 + l15) * LDSS + half * 8];
      af[mt].h[0] = *(const v8bf*)p;         // K = half*8 + 0..7
      af[mt].h[1] = *(const v8bf*)(p + 16);  // K = 16 + half*8 + 0..7
    }
#pragma unroll
    for (int nt = 0; nt < 4; ++nt) {
      BFrag bf;
      const __bf16* p = &bsb[(wn0 + nt * 16 + l15) * LDSS + half * 8];
      bf.h[0] = *(const v8bf*)p;  // K = half*8 + 0..7, 16 + half*8 + 0..7
      bf.h[1] = *(const v8bf*)(p + 16);
#pragma unroll
      for (int mt = 0; mt < 2; ++mt)
        acc[mt][nt] = wmma_bf16(af[mt].v, bf.v, acc[mt][nt]);
    }

    if (more) {
      store_tile(buf ^ 1, aN, bN);
      __syncthreads();
      buf ^= 1;
    }
  }

#pragma unroll
  for (int mt = 0; mt < 2; ++mt)
#pragma unroll
    for (int nt = 0; nt < 4; ++nt)
#pragma unroll
      for (int j = 0; j < 8; ++j) {
        int gr = m0 + wm0 + mt * 16 + half * 8 + j;
        int gc = n0 + wn0 + nt * 16 + l15;
        if (gr < Mdim && gc < Ndim) {
          float v = acc[mt][nt][j];
          if (bias) v += bias[gc];
          if (act == 1) v = 0.5f * v * (1.0f + erff(v * 0.70710678118654752f));
          long long off = coff + (long long)gr * ldcr + (long long)gc * ldcc;
          if (Cf) Cf[off] = v;
          if (Cb) Cb[off] = (__bf16)v;
        }
      }
}

// ---------------------------------------------------------------------------
// FAVOR+ feature kernel: per (b,h), tile of 64 rows x 256 features, K = 64.
// MODE 0: Q features (row max).  MODE 1: K global-max pass (atomic max).
// MODE 2: K features (uses kmax, accumulates ksum columns).
// ---------------------------------------------------------------------------
template <int MODE>
__global__ __launch_bounds__(256) void favor_kernel(
    const __bf16* __restrict__ QKVb, const __bf16* __restrict__ projb,
    int qkoff, __bf16* __restrict__ outP, unsigned* __restrict__ kmaxu,
    float* __restrict__ ksum) {
  constexpr int LDSS = 72;  // 64 + 8 pad; row pitch 144B (16B aligned)
  __shared__ __align__(16) __bf16 As[64 * LDSS];
  __shared__ __align__(16) __bf16 Bt[256 * LDSS];
  __shared__ float sdiag[64];
  __shared__ unsigned rowred[64];
  __shared__ float colsum[256];
  __shared__ unsigned smax;

  const int tid = threadIdx.x, lane = tid & 31, wave = tid >> 5;
  const int wm = wave >> 1, wn = wave & 1;
  const int half = lane >> 4, l15 = lane & 15;
  const int n0 = blockIdx.x * 64;
  const int bh = blockIdx.y;
  const int b = bh >> 4, h = bh & 15;

  if (MODE == 0 && tid < 64) rowred[tid] = fkey(-3.0e38f);
  if (MODE == 1 && tid == 0) smax = fkey(-3.0e38f);
  if (MODE == 2 && tid < 256) colsum[tid] = 0.0f;

  const __bf16* Ag = QKVb + (long long)(b * 4096 + n0) * 3072 + qkoff + h * 64;
#pragma unroll
  for (int t = 0; t < 2; ++t) {  // 64x64 tile: 512 groups of 8 (b128)
    int i = tid + t * 256;
    int r = i >> 3, c0 = (i & 7) * 8;
    *(v8bf*)&As[r * LDSS + c0] = *(const v8bf*)(Ag + (long long)r * 3072 + c0);
  }
#pragma unroll
  for (int t = 0; t < 8; ++t) {  // 256x64: 2048 groups of 8 (b128)
    int i = tid + t * 256;
    int m = i >> 3, c0 = (i & 7) * 8;
    *(v8bf*)&Bt[m * LDSS + c0] = *(const v8bf*)(projb + m * 64 + c0);
  }
  __syncthreads();

  if (MODE != 1 && tid < 64) {  // diag = 0.5 * HD^-0.5 * sum(q^2)
    float s = 0.0f;
#pragma unroll 8
    for (int k = 0; k < 64; ++k) {
      float q = (float)As[tid * LDSS + k];
      s += q * q;
    }
    sdiag[tid] = 0.0625f * s;
  }

  v8f acc[8];
#pragma unroll
  for (int nt = 0; nt < 8; ++nt) acc[nt] = vzero8();

#pragma unroll
  for (int ks = 0; ks < 2; ++ks) {
    const int kbase = ks * 32;
    BFrag af;
    {
      const __bf16* p = &As[(wm * 16 + l15) * LDSS + kbase + half * 8];
      af.h[0] = *(const v8bf*)p;
      af.h[1] = *(const v8bf*)(p + 16);
    }
#pragma unroll
    for (int nt = 0; nt < 8; ++nt) {
      BFrag bf;
      const __bf16* p =
          &Bt[(wn * 128 + nt * 16 + l15) * LDSS + kbase + half * 8];
      bf.h[0] = *(const v8bf*)p;
      bf.h[1] = *(const v8bf*)(p + 16);
      acc[nt] = wmma_bf16(af.v, bf.v, acc[nt]);
    }
  }

  if (MODE == 0) {  // per-row max over 256 features (in-register + LDS)
#pragma unroll
    for (int j = 0; j < 8; ++j) {
      float mj = acc[0][j];
#pragma unroll
      for (int nt = 1; nt < 8; ++nt) mj = fmaxf(mj, acc[nt][j]);
      mj = fmaxf(mj, __shfl_xor(mj, 1, 32));
      mj = fmaxf(mj, __shfl_xor(mj, 2, 32));
      mj = fmaxf(mj, __shfl_xor(mj, 4, 32));
      mj = fmaxf(mj, __shfl_xor(mj, 8, 32));
      if (l15 == 0) atomicMax(&rowred[wm * 16 + half * 8 + j], fkey(mj));
    }
  } else if (MODE == 1) {  // global max over tile
    float m = -3.0e38f;
#pragma unroll
    for (int nt = 0; nt < 8; ++nt)
#pragma unroll
      for (int j = 0; j < 8; ++j) m = fmaxf(m, acc[nt][j]);
    m = fmaxf(m, __shfl_xor(m, 1, 32));
    m = fmaxf(m, __shfl_xor(m, 2, 32));
    m = fmaxf(m, __shfl_xor(m, 4, 32));
    m = fmaxf(m, __shfl_xor(m, 8, 32));
    m = fmaxf(m, __shfl_xor(m, 16, 32));
    if (lane == 0) atomicMax(&smax, fkey(m));
  }
  __syncthreads();

  if (MODE == 1) {
    if (tid == 0) atomicMax(&kmaxu[bh], smax);
    return;
  }

  const float kmx = (MODE == 2) ? funkey(kmaxu[bh]) : 0.0f;
#pragma unroll
  for (int nt = 0; nt < 8; ++nt) {
    const int col = wn * 128 + nt * 16 + l15;
    float s = 0.0f;
#pragma unroll
    for (int j = 0; j < 8; ++j) {
      const int r = wm * 16 + half * 8 + j;
      const float mx = (MODE == 0) ? funkey(rowred[r]) : kmx;
      const float val = 0.0625f * (expf(acc[nt][j] - sdiag[r] - mx) + 1.0e-4f);
      outP[((long long)bh * 4096 + n0 + r) * 256 + col] = (__bf16)val;
      s += val;
    }
    if (MODE == 2) {
      s += __shfl_xor(s, 16, 32);  // other half = rows 8..15, same column
      if (half == 0) atomicAdd(&colsum[col], s);
    }
  }
  if (MODE == 2) {
    __syncthreads();
    if (tid < 256) atomicAdd(&ksum[bh * 256 + tid], colsum[tid]);
  }
}

// ---------------------------------------------------------------------------
// Small helper kernels
// ---------------------------------------------------------------------------
__global__ void cast_bf16_kernel(const float* __restrict__ src,
                                 __bf16* __restrict__ dst, long long n,
                                 float scale) {
  long long i = (long long)blockIdx.x * blockDim.x + threadIdx.x;
  long long stride = (long long)gridDim.x * blockDim.x;
  for (; i < n; i += stride) dst[i] = (__bf16)(src[i] * scale);
}

// src[R][C] (f32, coalesced read) -> dst[C][R] (bf16, transposed)
__global__ void cast_transpose_bf16(const float* __restrict__ src,
                                    __bf16* __restrict__ dst, int R, int C) {
  long long i = (long long)blockIdx.x * 256 + threadIdx.x;
  if (i < (long long)R * C) {
    int r = (int)(i / C), c = (int)(i % C);
    dst[(long long)c * R + r] = (__bf16)src[i];
  }
}

// builds W_qkv^T : Wt[c][k] (c in 0..3071, k in 0..1023) + packed bias
__global__ void pack_qkv_kernel(const float* __restrict__ wq,
                                const float* __restrict__ wk,
                                const float* __restrict__ wv,
                                const float* __restrict__ bq,
                                const float* __restrict__ bk,
                                const float* __restrict__ bv,
                                __bf16* __restrict__ Wt,
                                float* __restrict__ bias) {
  long long i = (long long)blockIdx.x * 256 + threadIdx.x;
  if (i < 1024LL * 3072) {
    int k = (int)(i / 3072), c = (int)(i % 3072);
    float v = (c < 1024)   ? wq[k * 1024 + c]
              : (c < 2048) ? wk[k * 1024 + c - 1024]
                           : wv[k * 1024 + c - 2048];
    Wt[(long long)c * 1024 + k] = (__bf16)v;
  }
  if (i < 3072) {
    int c = (int)i;
    bias[c] = (c < 1024) ? bq[c] : (c < 2048) ? bk[c - 1024] : bv[c - 2048];
  }
}

__global__ void init_kernel(unsigned* __restrict__ kmaxu,
                            float* __restrict__ ksum) {
  int i = blockIdx.x * 256 + threadIdx.x;
  if (i < 64) kmaxu[i] = fkey(-3.0e38f);
  if (i < 64 * 256) ksum[i] = 0.0f;
}

__global__ __launch_bounds__(256) void denom_kernel(
    const __bf16* __restrict__ Qp, const float* __restrict__ ksum,
    float* __restrict__ denom) {
  long long i = (long long)blockIdx.x * 256 + threadIdx.x;  // bh*4096 + n
  long long bh = i >> 12;
  const v8bf* q8 = (const v8bf*)(Qp + i * 256);
  const float* ks = ksum + bh * 256;
  float s = 0.0f;
#pragma unroll 4
  for (int g = 0; g < 32; ++g) {
    v8bf v = q8[g];
#pragma unroll
    for (int e = 0; e < 8; ++e) s += (float)v[e] * ks[g * 8 + e];
  }
  denom[i] = s;
}

__global__ __launch_bounds__(256) void merge_kernel(
    const float* __restrict__ attnh, const float* __restrict__ denom,
    __bf16* __restrict__ out) {
  long long i = (long long)blockIdx.x * 256 + threadIdx.x;  // row*1024 + col
  int col = (int)(i & 1023);
  long long row = i >> 10;
  int h = col >> 6, d = col & 63;
  int b = (int)(row >> 12), n = (int)(row & 4095);
  long long bh = b * 16 + h;
  float v = attnh[(bh * 4096 + n) * 64 + d] / denom[bh * 4096 + n];
  out[i] = (__bf16)v;
}

__global__ __launch_bounds__(256) void ln_kernel(
    const float* __restrict__ y, const float* __restrict__ res,
    const float* __restrict__ g, const float* __restrict__ bta,
    float* __restrict__ outf, __bf16* __restrict__ outb) {
  const int row = blockIdx.x, tid = threadIdx.x;
  const float* yr = y + (long long)row * 1024;
  const float* rr = res + (long long)row * 1024;
  float v[4];
  float s = 0.0f;
#pragma unroll
  for (int i = 0; i < 4; ++i) {
    int c = tid * 4 + i;
    v[i] = yr[c] + rr[c];
    s += v[i];
  }
  __shared__ float red[256];
  red[tid] = s;
  __syncthreads();
  for (int off = 128; off > 0; off >>= 1) {
    if (tid < off) red[tid] += red[tid + off];
    __syncthreads();
  }
  float mu = red[0] * (1.0f / 1024.0f);
  __syncthreads();
  float s2 = 0.0f;
#pragma unroll
  for (int i = 0; i < 4; ++i) {
    float d = v[i] - mu;
    s2 += d * d;
  }
  red[tid] = s2;
  __syncthreads();
  for (int off = 128; off > 0; off >>= 1) {
    if (tid < off) red[tid] += red[tid + off];
    __syncthreads();
  }
  float rstd = rsqrtf(red[0] * (1.0f / 1024.0f) + 1e-5f);
#pragma unroll
  for (int i = 0; i < 4; ++i) {
    int c = tid * 4 + i;
    float o = (v[i] - mu) * rstd * g[c] + bta[c];
    if (outf) outf[(long long)row * 1024 + c] = o;
    if (outb) outb[(long long)row * 1024 + c] = (__bf16)o;
  }
}

// ---------------------------------------------------------------------------
static void launch_gemm(hipStream_t s, const __bf16* A, long long sAr,
                        long long sAc, long long sAb0, long long sAb1,
                        const __bf16* Bm, long long sBr, long long sBc,
                        long long sBb0, long long sBb1, const float* bias,
                        float* Cf, __bf16* Cb, long long ldcr, long long ldcc,
                        long long sCb0, long long sCb1, int M, int N, int K,
                        int batch, int hmod, int act) {
  dim3 grid((N + 127) / 128, (M + 127) / 128, batch);
  gemm_wmma<<<grid, 256, 0, s>>>(A, sAr, sAc, sAb0, sAb1, Bm, sBr, sBc, sBb0,
                                 sBb1, bias, Cf, Cb, ldcr, ldcc, sCb0, sCb1, M,
                                 N, K, hmod, act);
}

extern "C" void kernel_launch(void* const* d_in, const int* in_sizes, int n_in,
                              void* d_out, int out_size, void* d_ws,
                              size_t ws_size, hipStream_t stream) {
  (void)in_sizes; (void)n_in; (void)out_size; (void)ws_size;
  const float* x    = (const float*)d_in[0];
  const float* wq   = (const float*)d_in[1];
  const float* bq   = (const float*)d_in[2];
  const float* wk   = (const float*)d_in[3];
  const float* bk   = (const float*)d_in[4];
  const float* wv   = (const float*)d_in[5];
  const float* bv   = (const float*)d_in[6];
  const float* wo   = (const float*)d_in[7];
  const float* bo   = (const float*)d_in[8];
  const float* proj = (const float*)d_in[9];
  const float* w1   = (const float*)d_in[10];
  const float* b1   = (const float*)d_in[11];
  const float* w2   = (const float*)d_in[12];
  const float* b2   = (const float*)d_in[13];
  const float* g1   = (const float*)d_in[14];
  const float* be1  = (const float*)d_in[15];
  const float* g2   = (const float*)d_in[16];
  const float* be2  = (const float*)d_in[17];
  float* out = (float*)d_out;

  char* ws = (char*)d_ws;
  size_t o = 0;
  auto alloc = [&](size_t bytes) -> void* {
    o = (o + 255) & ~(size_t)255;
    void* p = ws + o;
    o += bytes;
    return p;
  };
  __bf16* xb     = (__bf16*)alloc(16384LL * 1024 * 2);
  __bf16* Wqkvt  = (__bf16*)alloc(3072LL * 1024 * 2);   // [3072][1024] = W^T
  float*  bqkv   = (float*)alloc(3072 * 4);
  __bf16* wo_t   = (__bf16*)alloc(1024LL * 1024 * 2);   // [N=1024][K=1024]
  __bf16* w1_t   = (__bf16*)alloc(4096LL * 1024 * 2);   // [N=4096][K=1024]
  __bf16* w2_t   = (__bf16*)alloc(1024LL * 4096 * 2);   // [N=1024][K=4096]
  __bf16* projb  = (__bf16*)alloc(256LL * 64 * 2);
  __bf16* QKVb   = (__bf16*)alloc(16384LL * 3072 * 2);
  __bf16* Qp     = (__bf16*)alloc(64LL * 4096 * 256 * 2);  // reused: FFN hidden
  __bf16* Kp     = (__bf16*)alloc(64LL * 4096 * 256 * 2);  // reused: ybuf + x1
  unsigned* kmaxu = (unsigned*)alloc(64 * 4);
  float* ksum    = (float*)alloc(64LL * 256 * 4);
  __bf16* kvbT   = (__bf16*)alloc(64LL * 64 * 256 * 2);    // [bh][d][m]
  float* attnh   = (float*)alloc(64LL * 4096 * 64 * 4);    // reused: x1b
  float* denom   = (float*)alloc(64LL * 4096 * 4);
  __bf16* attn_b = (__bf16*)alloc(16384LL * 1024 * 2);
  // aliases (lifetimes verified: source buffer dead before reuse)
  __bf16* hb  = Qp;
  float* ybuf = (float*)Kp;
  float* x1   = (float*)((char*)Kp + 16384LL * 1024 * 4);
  __bf16* x1b = (__bf16*)attnh;

  // --- pack / cast (weights transposed so GEMM B-operand is K-contiguous) ---
  cast_bf16_kernel<<<4096, 256, 0, stream>>>(x, xb, 16384LL * 1024, 1.0f);
  pack_qkv_kernel<<<12288, 256, 0, stream>>>(wq, wk, wv, bq, bk, bv, Wqkvt, bqkv);
  cast_transpose_bf16<<<4096, 256, 0, stream>>>(wo, wo_t, 1024, 1024);
  cast_transpose_bf16<<<16384, 256, 0, stream>>>(w1, w1_t, 1024, 4096);
  cast_transpose_bf16<<<16384, 256, 0, stream>>>(w2, w2_t, 4096, 1024);
  cast_bf16_kernel<<<64, 256, 0, stream>>>(proj, projb, 256LL * 64,
                                           0.3535533905932738f);  // 64^-0.25
  init_kernel<<<64, 256, 0, stream>>>(kmaxu, ksum);

  // --- fused QKV projection: [16384,1024] x [1024,3072] ---
  launch_gemm(stream, xb, 1024, 1, 0, 0, Wqkvt, 1, 1024, 0, 0, bqkv, nullptr,
              QKVb, 3072, 1, 0, 0, 16384, 3072, 1024, 1, 1, 0);

  // --- FAVOR+ features (64 heads, N=4096, M=256, K=64) ---
  favor_kernel<0><<<dim3(64, 64), 256, 0, stream>>>(QKVb, projb, 0, Qp, kmaxu,
                                                    ksum);
  favor_kernel<1><<<dim3(64, 64), 256, 0, stream>>>(QKVb, projb, 1024, nullptr,
                                                    kmaxu, ksum);
  favor_kernel<2><<<dim3(64, 64), 256, 0, stream>>>(QKVb, projb, 1024, Kp,
                                                    kmaxu, ksum);

  // --- kv = Kp^T V  (batched 64: M=256, N=64, K=4096; A transposed via
  //     strides, C stored transposed -> kvbT[bh][d][m]) ---
  launch_gemm(stream, Kp, 1, 256, 16LL * 1048576, 1048576, QKVb + 2048, 3072, 1,
              4096LL * 3072, 64, nullptr, nullptr, kvbT, 1, 256, 16LL * 16384,
              16384, 256, 64, 4096, 64, 16, 0);
  // --- attn = Qp kv  (batched 64: M=4096, N=64, K=256; B = kvbT, K-contig) ---
  launch_gemm(stream, Qp, 256, 1, 1048576, 0, kvbT, 1, 256, 16384, 0, nullptr,
              attnh, nullptr, 64, 1, 262144, 0, 4096, 64, 256, 64, 1, 0);
  denom_kernel<<<1024, 256, 0, stream>>>(Qp, ksum, denom);
  merge_kernel<<<65536, 256, 0, stream>>>(attnh, denom, attn_b);

  // --- output projection + residual LN ---
  launch_gemm(stream, attn_b, 1024, 1, 0, 0, wo_t, 1, 1024, 0, 0, bo, ybuf,
              nullptr, 1024, 1, 0, 0, 16384, 1024, 1024, 1, 1, 0);
  ln_kernel<<<16384, 256, 0, stream>>>(ybuf, x, g1, be1, x1, x1b);

  // --- FFN: GELU(x1 W1 + b1) W2 + b2, residual LN ---
  launch_gemm(stream, x1b, 1024, 1, 0, 0, w1_t, 1, 1024, 0, 0, b1, nullptr, hb,
              4096, 1, 0, 0, 16384, 4096, 1024, 1, 1, 1);
  launch_gemm(stream, hb, 4096, 1, 0, 0, w2_t, 1, 4096, 0, 0, b2, ybuf, nullptr,
              1024, 1, 0, 0, 16384, 1024, 4096, 1, 1, 0);
  ln_kernel<<<16384, 256, 0, stream>>>(ybuf, x1, g2, be2, out, nullptr);
}
